// EnergyTransformer_9835475107941
// MI455X (gfx1250) — matrix operand order repaired and checked by reading, not compiled, not run on hardware
//
#include <hip/hip_runtime.h>
#include <hip/hip_bf16.h>

typedef __bf16 bf16_t;
typedef __attribute__((ext_vector_type(16))) __bf16 v16bf;
typedef __attribute__((ext_vector_type(8)))  __bf16 v8bf;
typedef __attribute__((ext_vector_type(8)))  float  v8f;

namespace {

constexpr int  B_    = 4;
constexpr int  N_    = 512;
constexpr int  D_    = 768;   // == H_*HD_
constexpr int  H_    = 12;
constexpr int  HD_   = 64;
constexpr int  M_    = 3072;
constexpr int  ROWS_ = B_ * N_;          // 2048
constexpr int  STEPS_ = 12;
constexpr float ALPHA_ = 0.125f;
constexpr float EPS_   = 1e-5f;
constexpr float BETA_  = 0.125f;         // 1/sqrt(64)

// ---------------------------------------------------------------------------
// Contiguous-K fragment loads (CDNA5 VGPR layouts, CDNA5 ISA):
// A (16x32): lane row = lane&15; lanes>=16 use K-base 8. Lane data = two
//            contiguous 8-element (16B) runs at K = kb and kb+16.
// B (32x16): lane col = lane&15; lanes>=16 use K-base 8. Lane data = two
//            contiguous 8-element (16B) runs at K = kb and kb+16.
// ---------------------------------------------------------------------------
__device__ inline v16bf ld_frag_a(const bf16_t* __restrict__ p) {
  v8bf lo = *(const v8bf*)(p);        // K = kb .. kb+7
  v8bf hi = *(const v8bf*)(p + 16);   // K = kb+16 .. kb+23
  return __builtin_shufflevector(lo, hi, 0,1,2,3,4,5,6,7,8,9,10,11,12,13,14,15);
}
__device__ inline v16bf ld_frag_b(const bf16_t* __restrict__ p) {
  v8bf lo = *(const v8bf*)(p);        // K = kb .. kb+7
  v8bf hi = *(const v8bf*)(p + 16);   // K = kb+16 .. kb+23
  return __builtin_shufflevector(lo, hi, 0,1,2,3,4,5,6,7,8,9,10,11,12,13,14,15);
}

// ---------------------------------------------------------------------------
// Batched bf16 WMMA GEMM:  C[z] = scale * op( A[z] @ B[z] )
// Both operands K-contiguous (stride-1 in K). Block = 128 thr = 4 waves,
// each wave computes a 64x32 C tile (4x2 frags, 8 WMMAs / K=32 slice).
// Block tile 128x64; grid (M/128, N/64, Z), z = (zo,zi) via Hi.
// Base pointers are loop-invariant scalars (indexed by kt) so clang's
// address-space inference keeps the loads in the GLOBAL path (loadcnt only),
// not FLAT (loadcnt+dscnt).
// flags: 1 -> store bf16 (Cb), 2 -> relu before scale, 4 -> accumulate (Cf)
// ---------------------------------------------------------------------------
__global__ __launch_bounds__(128)
void wmma_gemm_bf16(const bf16_t* __restrict__ A, const bf16_t* __restrict__ B,
                    float* __restrict__ Cf, bf16_t* __restrict__ Cb,
                    int K, int Hi,
                    int a_sm, int a_sbo, int a_sbi,
                    int b_sn, int b_sbo, int b_sbi,
                    int c_sm, int c_sbo, int c_sbi,
                    float scale, int flags)
{
  int z  = blockIdx.z;
  int zo = z / Hi, zi = z - zo * Hi;
  const bf16_t* __restrict__ Ab = A + (long)zo * a_sbo + (long)zi * a_sbi;
  const bf16_t* __restrict__ Bb = B + (long)zo * b_sbo + (long)zi * b_sbi;
  int coff = zo * c_sbo + zi * c_sbi;

  int lane = threadIdx.x & 31;
  int w    = threadIdx.x >> 5;
  int m0   = blockIdx.x * 128 + (w >> 1) * 64;
  int n0   = blockIdx.y * 64  + (w & 1) * 32;
  int l15  = lane & 15;
  int kbA  = (lane & 16) ? 8  : 0;
  int kbB  = (lane & 16) ? 8  : 0;

  const bf16_t* __restrict__ a0 = Ab + (long)(m0 + l15) * a_sm + kbA;
  const bf16_t* __restrict__ a1 = a0 + 16 * a_sm;
  const bf16_t* __restrict__ a2 = a0 + 32 * a_sm;
  const bf16_t* __restrict__ a3 = a0 + 48 * a_sm;
  const bf16_t* __restrict__ b0 = Bb + (long)(n0 + l15) * b_sn + kbB;
  const bf16_t* __restrict__ b1 = b0 + 16 * b_sn;

  v8f acc[4][2] = {};

  for (int kt = 0; kt < K; kt += 32) {
    v16bf fa0 = ld_frag_a(a0 + kt);
    v16bf fa1 = ld_frag_a(a1 + kt);
    v16bf fa2 = ld_frag_a(a2 + kt);
    v16bf fa3 = ld_frag_a(a3 + kt);
    v16bf fb0 = ld_frag_b(b0 + kt);
    v16bf fb1 = ld_frag_b(b1 + kt);
    acc[0][0] = __builtin_amdgcn_wmma_f32_16x16x32_bf16(false, fa0, false, fb0, (short)0, acc[0][0], false, false);
    acc[0][1] = __builtin_amdgcn_wmma_f32_16x16x32_bf16(false, fa0, false, fb1, (short)0, acc[0][1], false, false);
    acc[1][0] = __builtin_amdgcn_wmma_f32_16x16x32_bf16(false, fa1, false, fb0, (short)0, acc[1][0], false, false);
    acc[1][1] = __builtin_amdgcn_wmma_f32_16x16x32_bf16(false, fa1, false, fb1, (short)0, acc[1][1], false, false);
    acc[2][0] = __builtin_amdgcn_wmma_f32_16x16x32_bf16(false, fa2, false, fb0, (short)0, acc[2][0], false, false);
    acc[2][1] = __builtin_amdgcn_wmma_f32_16x16x32_bf16(false, fa2, false, fb1, (short)0, acc[2][1], false, false);
    acc[3][0] = __builtin_amdgcn_wmma_f32_16x16x32_bf16(false, fa3, false, fb0, (short)0, acc[3][0], false, false);
    acc[3][1] = __builtin_amdgcn_wmma_f32_16x16x32_bf16(false, fa3, false, fb1, (short)0, acc[3][1], false, false);
  }

  int rb = (lane & 16) ? 8 : 0;
#pragma unroll
  for (int i = 0; i < 4; ++i)
#pragma unroll
    for (int j = 0; j < 2; ++j) {
      int nIdx = coff + n0 + 16 * j + l15;
#pragma unroll
      for (int r = 0; r < 8; ++r) {
        float v = acc[i][j][r];
        if (flags & 2) v = v > 0.f ? v : 0.f;
        v *= scale;
        long idx = (long)nIdx + (long)(m0 + 16 * i + rb + r) * c_sm;
        if (flags & 1) {
          Cb[idx] = (bf16_t)v;
        } else {
          if (flags & 4) v += Cf[idx];
          Cf[idx] = v;
        }
      }
    }
}

// ---------------------------------------------------------------------------
// Batched bf16 transpose via LDS 32x33 tile (coalesced read + write).
// in(z, r, c) = in + zo*sbo + zi*sbi + r*si + c ;  out[z][c][r] contiguous.
// grid (R/32, C/32, Z), block (32, 8).
// ---------------------------------------------------------------------------
__global__ __launch_bounds__(256)
void transpose_bf(const bf16_t* __restrict__ in, bf16_t* __restrict__ out,
                  int R, int C, int Hi, int si, int sbo, int sbi)
{
  __shared__ bf16_t tile[32][33];
  int z = blockIdx.z;
  int zo = z / Hi, zi = z - zo * Hi;
  const bf16_t* __restrict__ ib = in + (long)zo * sbo + (long)zi * sbi;
  bf16_t* __restrict__ ob = out + (long)z * R * C;
  int r0 = blockIdx.x * 32, c0 = blockIdx.y * 32;
  int tx = threadIdx.x, ty = threadIdx.y;
#pragma unroll
  for (int yy = 0; yy < 4; ++yy) {
    int r = ty + 8 * yy;
    tile[r][tx] = ib[(long)(r0 + r) * si + (c0 + tx)];
  }
  __syncthreads();
#pragma unroll
  for (int yy = 0; yy < 4; ++yy) {
    int c = ty + 8 * yy;
    ob[(long)(c0 + c) * R + (r0 + tx)] = tile[tx][c];
  }
}

// ---------------------------------------------------------------------------
// LayerNorm forward: g = gamma * (x - mu) * rsqrt(var + eps) + delta  (-> bf16)
// ---------------------------------------------------------------------------
__global__ __launch_bounds__(256)
void ln_fwd(const float* __restrict__ x, const float* __restrict__ gamma,
            const float* __restrict__ delta, bf16_t* __restrict__ g)
{
  __shared__ float red[256];
  long row = blockIdx.x;
  const float* xr = x + row * D_;
  int t = threadIdx.x;
  float xv[3];
  float s = 0.f;
#pragma unroll
  for (int i = 0; i < 3; ++i) { xv[i] = xr[t + 256 * i]; s += xv[i]; }
  red[t] = s; __syncthreads();
  for (int o = 128; o > 0; o >>= 1) { if (t < o) red[t] += red[t + o]; __syncthreads(); }
  float mu = red[0] * (1.f / D_);
  __syncthreads();
  float v = 0.f;
#pragma unroll
  for (int i = 0; i < 3; ++i) { float d = xv[i] - mu; v += d * d; }
  red[t] = v; __syncthreads();
  for (int o = 128; o > 0; o >>= 1) { if (t < o) red[t] += red[t + o]; __syncthreads(); }
  float rsig = rsqrtf(red[0] * (1.f / D_) + EPS_);
#pragma unroll
  for (int i = 0; i < 3; ++i) {
    int d = t + 256 * i;
    g[row * D_ + d] = (bf16_t)(gamma[d] * (xv[i] - mu) * rsig + delta[d]);
  }
}

// ---------------------------------------------------------------------------
// In-place softmax over rows of 512 (bf16 -> bf16), one block per row.
// ---------------------------------------------------------------------------
__global__ __launch_bounds__(256)
void softmax_rows(bf16_t* __restrict__ p)
{
  __shared__ float red[256];
  long row = blockIdx.x;
  bf16_t* pr = p + row * (long)N_;
  int t = threadIdx.x;
  float v0 = (float)pr[t], v1 = (float)pr[t + 256];
  red[t] = fmaxf(v0, v1); __syncthreads();
  for (int o = 128; o > 0; o >>= 1) { if (t < o) red[t] = fmaxf(red[t], red[t + o]); __syncthreads(); }
  float mx = red[0];
  __syncthreads();
  float e0 = __expf(v0 - mx), e1 = __expf(v1 - mx);
  red[t] = e0 + e1; __syncthreads();
  for (int o = 128; o > 0; o >>= 1) { if (t < o) red[t] += red[t + o]; __syncthreads(); }
  float inv = 1.f / red[0];
  pr[t]       = (bf16_t)(e0 * inv);
  pr[t + 256] = (bf16_t)(e1 * inv);
}

// ---------------------------------------------------------------------------
// LayerNorm backward + gradient-descent update (f32, in place on x).
// ---------------------------------------------------------------------------
__global__ __launch_bounds__(256)
void ln_bwd_update(float* __restrict__ x, const float* __restrict__ gg,
                   const float* __restrict__ gamma)
{
  __shared__ float red[256];
  long row = blockIdx.x;
  float* xr = x + row * D_;
  const float* gr = gg + row * D_;
  int t = threadIdx.x;
  float xv[3], tv[3];
  float s = 0.f;
#pragma unroll
  for (int i = 0; i < 3; ++i) { xv[i] = xr[t + 256 * i]; s += xv[i]; }
  red[t] = s; __syncthreads();
  for (int o = 128; o > 0; o >>= 1) { if (t < o) red[t] += red[t + o]; __syncthreads(); }
  float mu = red[0] * (1.f / D_);
  __syncthreads();
  float v = 0.f;
#pragma unroll
  for (int i = 0; i < 3; ++i) { float d = xv[i] - mu; v += d * d; }
  red[t] = v; __syncthreads();
  for (int o = 128; o > 0; o >>= 1) { if (t < o) red[t] += red[t + o]; __syncthreads(); }
  float rsig = rsqrtf(red[0] * (1.f / D_) + EPS_);
  __syncthreads();
  float s1 = 0.f, s2 = 0.f;
#pragma unroll
  for (int i = 0; i < 3; ++i) {
    int d = t + 256 * i;
    tv[i] = gr[d] * gamma[d];
    float xh = (xv[i] - mu) * rsig;
    s1 += tv[i];
    s2 += tv[i] * xh;
  }
  red[t] = s1; __syncthreads();
  for (int o = 128; o > 0; o >>= 1) { if (t < o) red[t] += red[t + o]; __syncthreads(); }
  s1 = red[0] * (1.f / D_);
  __syncthreads();
  red[t] = s2; __syncthreads();
  for (int o = 128; o > 0; o >>= 1) { if (t < o) red[t] += red[t + o]; __syncthreads(); }
  s2 = red[0] * (1.f / D_);
#pragma unroll
  for (int i = 0; i < 3; ++i) {
    float xh = (xv[i] - mu) * rsig;
    float dx = (tv[i] - s1 - xh * s2) * rsig;
    xr[t + 256 * i] = xv[i] - ALPHA_ * dx;
  }
}

// ---------------------------------------------------------------------------
// Weight prep (once per launch): W[H,D,HD] f32 ->
//   wb [d][h*HD+e] bf16  (K-major B for the grad backprojection)
//   wbt[h*HD+e][d] bf16  (K-major B for the forward projection)
// ---------------------------------------------------------------------------
__global__ __launch_bounds__(256)
void pack_w(const float* __restrict__ W, bf16_t* __restrict__ wb, bf16_t* __restrict__ wbt)
{
  long i = (long)blockIdx.x * 256 + threadIdx.x;
  if (i >= (long)H_ * D_ * HD_) return;
  int e = (int)(i % HD_);
  long r = i / HD_;
  int d = (int)(r % D_);
  int h = (int)(r / D_);
  bf16_t v = (bf16_t)W[i];
  int c = h * HD_ + e;
  wb [(long)d * D_ + c] = v;
  wbt[(long)c * D_ + d] = v;
}

__global__ __launch_bounds__(256)
void prep_xi(const float* __restrict__ xi, bf16_t* __restrict__ xib, bf16_t* __restrict__ xit)
{
  long i = (long)blockIdx.x * 256 + threadIdx.x;
  if (i >= (long)M_ * D_) return;
  int d = (int)(i % D_);
  long m = i / D_;
  bf16_t v = (bf16_t)xi[i];
  xib[i] = v;
  xit[(long)d * M_ + m] = v;
}

inline void gemm(hipStream_t st, const bf16_t* A, const bf16_t* B,
                 float* Cf, bf16_t* Cb, int M, int N, int K, int Z, int Hi,
                 int a_sm, int a_sbo, int a_sbi,
                 int b_sn, int b_sbo, int b_sbi,
                 int c_sm, int c_sbo, int c_sbi,
                 float scale, int flags)
{
  dim3 g(M / 128, N / 64, Z), blk(128);
  wmma_gemm_bf16<<<g, blk, 0, st>>>(A, B, Cf, Cb, K, Hi,
                                    a_sm, a_sbo, a_sbi,
                                    b_sn, b_sbo, b_sbi,
                                    c_sm, c_sbo, c_sbi,
                                    scale, flags);
}

} // namespace

extern "C" void kernel_launch(void* const* d_in, const int* in_sizes, int n_in,
                              void* d_out, int out_size, void* d_ws, size_t ws_size,
                              hipStream_t stream)
{
  (void)in_sizes; (void)n_in; (void)out_size;
  const float* x_in  = (const float*)d_in[0];
  const float* gamma = (const float*)d_in[1];
  const float* delta = (const float*)d_in[2];
  const float* Wq    = (const float*)d_in[3];
  const float* Wk    = (const float*)d_in[4];
  const float* xi    = (const float*)d_in[5];

  // -------- workspace carving (256B aligned) --------
  char* p = (char*)d_ws;
  auto take = [&p](size_t bytes) -> char* {
    char* r = p; p += (bytes + 255) & ~(size_t)255; return r;
  };
  float*  xcur   = (float*) take((size_t)ROWS_ * D_ * 4);
  bf16_t* gbf    = (bf16_t*)take((size_t)ROWS_ * D_ * 2);
  bf16_t* wq_bf  = (bf16_t*)take((size_t)D_ * D_ * 2);
  bf16_t* wqT_bf = (bf16_t*)take((size_t)D_ * D_ * 2);
  bf16_t* wk_bf  = (bf16_t*)take((size_t)D_ * D_ * 2);
  bf16_t* wkT_bf = (bf16_t*)take((size_t)D_ * D_ * 2);
  bf16_t* xi_bf  = (bf16_t*)take((size_t)M_ * D_ * 2);
  bf16_t* xiT_bf = (bf16_t*)take((size_t)D_ * M_ * 2);
  bf16_t* q_bf   = (bf16_t*)take((size_t)ROWS_ * D_ * 2);
  bf16_t* k_bf   = (bf16_t*)take((size_t)ROWS_ * D_ * 2);
  bf16_t* qT_bf  = (bf16_t*)take((size_t)B_ * H_ * HD_ * N_ * 2); // [z][64][512]
  bf16_t* kT_bf  = (bf16_t*)take((size_t)B_ * H_ * HD_ * N_ * 2);
  bf16_t* sbuf   = (bf16_t*)take((size_t)B_ * H_ * N_ * N_ * 2);  // scores -> P
  bf16_t* pT_bf  = (bf16_t*)take((size_t)B_ * H_ * N_ * N_ * 2);  // P^T
  bf16_t* dq_bf  = (bf16_t*)take((size_t)ROWS_ * D_ * 2);
  bf16_t* dk_bf  = (bf16_t*)take((size_t)ROWS_ * D_ * 2);
  bf16_t* hrelu  = (bf16_t*)take((size_t)ROWS_ * M_ * 2);
  float*  gradg  = (float*) take((size_t)ROWS_ * D_ * 4);
  if ((size_t)(p - (char*)d_ws) > ws_size) return;  // scratch too small; bail deterministically

  // -------- one-time prep --------
  hipMemcpyAsync(xcur, x_in, (size_t)ROWS_ * D_ * 4, hipMemcpyDeviceToDevice, stream);
  {
    long nW = (long)H_ * D_ * HD_;                 // 589824
    pack_w<<<dim3((nW + 255) / 256), dim3(256), 0, stream>>>(Wq, wq_bf, wqT_bf);
    pack_w<<<dim3((nW + 255) / 256), dim3(256), 0, stream>>>(Wk, wk_bf, wkT_bf);
    long nX = (long)M_ * D_;                       // 2359296
    prep_xi<<<dim3((nX + 255) / 256), dim3(256), 0, stream>>>(xi, xi_bf, xiT_bf);
  }

  const int BH   = 512 * 768;        // per-b stride in q/k/dq/dk
  const int SBb  = 12 * 512 * 512;   // per-b stride in score buffer
  const int SBh  = 512 * 512;        // per-h stride in score buffer
  const int TQz  = 64 * 512;         // per-z stride in qT/kT  ([z][64][512])
  const int PTz  = 512 * 512;        // per-z stride in pT

  for (int s = 0; s < STEPS_; ++s) {
    // g = LN(x)  (bf16)
    ln_fwd<<<dim3(ROWS_), dim3(256), 0, stream>>>(xcur, gamma, delta, gbf);

    // Q = g @ Wq ;  K = g @ Wk
    gemm(stream, gbf, wqT_bf, nullptr, q_bf, ROWS_, D_, D_, 1, 1,
         768, 0, 0,   768, 0, 0,   768, 0, 0,   1.f, 1);
    gemm(stream, gbf, wkT_bf, nullptr, k_bf, ROWS_, D_, D_, 1, 1,
         768, 0, 0,   768, 0, 0,   768, 0, 0,   1.f, 1);

    // S = beta * Q Kt per (b,h)
    gemm(stream, q_bf, k_bf, nullptr, sbuf, 512, 512, 64, B_ * H_, H_,
         768, BH, 64,   768, BH, 64,   512, SBb, SBh,   BETA_, 1);

    // P = softmax(S), in place
    softmax_rows<<<dim3(B_ * H_ * N_), dim3(256), 0, stream>>>(sbuf);

    // Transposes to keep GEMM operands K-contiguous
    transpose_bf<<<dim3(16, 2, B_ * H_), dim3(32, 8), 0, stream>>>(
        q_bf, qT_bf, 512, 64, H_, 768, BH, 64);
    transpose_bf<<<dim3(16, 2, B_ * H_), dim3(32, 8), 0, stream>>>(
        k_bf, kT_bf, 512, 64, H_, 768, BH, 64);
    transpose_bf<<<dim3(16, 16, B_ * H_), dim3(32, 8), 0, stream>>>(
        sbuf, pT_bf, 512, 512, H_, 512, SBb, SBh);

    // dQ = -(P @ K)
    gemm(stream, sbuf, kT_bf, nullptr, dq_bf, 512, 64, 512, B_ * H_, H_,
         512, SBb, SBh,   512, H_ * TQz, TQz,   768, BH, 64,   -1.f, 1);

    // dK = -(Pt @ Q)
    gemm(stream, pT_bf, qT_bf, nullptr, dk_bf, 512, 64, 512, B_ * H_, H_,
         512, H_ * PTz, PTz,   512, H_ * TQz, TQz,   768, BH, 64,   -1.f, 1);

    // Hrelu = relu(g @ xiT)
    gemm(stream, gbf, xi_bf, nullptr, hrelu, ROWS_, M_, D_, 1, 1,
         768, 0, 0,   768, 0, 0,   3072, 0, 0,   1.f, 1 | 2);

    // grad_g  = -(Hrelu @ xi)
    gemm(stream, hrelu, xiT_bf, gradg, nullptr, ROWS_, D_, M_, 1, 1,
         3072, 0, 0,   3072, 0, 0,   768, 0, 0,   -1.f, 0);
    // grad_g += dQ @ WqT
    gemm(stream, dq_bf, wq_bf, gradg, nullptr, ROWS_, D_, D_, 1, 1,
         768, 0, 0,   768, 0, 0,   768, 0, 0,   1.f, 4);
    // grad_g += dK @ WkT
    gemm(stream, dk_bf, wk_bf, gradg, nullptr, ROWS_, D_, D_, 1, 1,
         768, 0, 0,   768, 0, 0,   768, 0, 0,   1.f, 4);

    // LN backward + x update (in place)
    ln_bwd_update<<<dim3(ROWS_), dim3(256), 0, stream>>>(xcur, gradg, gamma);
  }

  hipMemcpyAsync(d_out, xcur, (size_t)ROWS_ * D_ * 4, hipMemcpyDeviceToDevice, stream);
}
